// QRNN_8186207666207
// MI455X (gfx1250) — matrix operand order repaired and checked by reading, not compiled, not run on hardware
//
#include <hip/hip_runtime.h>
#include <hip/hip_bf16.h>

typedef __bf16 bf16_t;
typedef bf16_t v8bf  __attribute__((ext_vector_type(8)));
typedef bf16_t v16bf __attribute__((ext_vector_type(16)));
typedef float  v8f   __attribute__((ext_vector_type(8)));

#define B_      64
#define S_      2048
#define H_      256
#define K_      128
#define EMB_    124
#define VOCAB_  128
#define TCH     32            // timesteps per chunk
#define NCH     (S_/TCH)      // 64 chunks
#define JBLK    64            // hidden units per workgroup
#define NJB     (H_/JBLK)     // 4 j-blocks
#define CSTRIDE 136           // comb row stride (bf16 elems); 272B = 17*16B (aligned, bank-rotating)
#define GSTR    140           // gate row stride (floats): 560B -> t/t+8 rows land 32 banks apart

__device__ __forceinline__ float fast_sigmoid(float x) {
    return __builtin_amdgcn_rcpf(1.0f + __expf(-x));
}
__device__ __forceinline__ float fast_tanh(float x) {
    return 1.0f - 2.0f * __builtin_amdgcn_rcpf(1.0f + __expf(2.0f * x));
}

__global__ __launch_bounds__(256)
void qrnn_fused(const float* __restrict__ X,     // (B,S,8)
                const float* __restrict__ emb,   // (128,124)
                const float* __restrict__ Wnum,  // (4,7)
                const float* __restrict__ bnum,  // (4)
                const float* __restrict__ Wconv, // (768,128)
                const float* __restrict__ bconv, // (768)
                const float* __restrict__ Wout,  // (256)
                float* __restrict__ out)         // (B,1)
{
    __shared__ bf16_t embS[VOCAB_][128];      // 32 KB, cols 124..127 = 0
    __shared__ float  Xc[TCH][8];             // raw X chunk
    __shared__ bf16_t comb[TCH][CSTRIDE];     // A-matrix (bf16)
    __shared__ float  gZF[TCH][GSTR];         // interleaved gates: [t][2*j + (0:Z,1:F)]
    __shared__ float  ACbuf[4][JBLK][2];      // per-quarter affine (A,C)
    __shared__ float  wnumS[4][7];
    __shared__ float  bnumS[4];
    __shared__ float  red[64];

    const int b    = blockIdx.x;
    const int j0   = blockIdx.y * JBLK;
    const int tid  = threadIdx.x;
    const int lane = tid & 31;
    const int wv   = tid >> 5;     // 0..7 (8 waves)
    const int g    = wv >> 2;      // 0 = Z tile, 1 = F tile
    const int nt   = wv & 3;       // N-tile within j-block
    const int ncol = lane & 15;    // N within tile / C-matrix column
    const int hi   = lane >> 4;    // half-wave select

    // ---- one-time staging ----------------------------------------------
    for (int idx = tid; idx < VOCAB_ * 128; idx += 256) {
        int r = idx >> 7, c = idx & 127;
        embS[r][c] = (c < EMB_) ? (bf16_t)emb[r * EMB_ + c] : (bf16_t)0.0f;
    }
    if (tid < 28) ((float*)wnumS)[tid] = Wnum[tid];
    if (tid < 4)  bnumS[tid] = bnum[tid];

    // Loop-invariant B fragments (ISA B layout: lanes 0-15 K 0..7,16..23,
    // lanes 16-31 K 8..15,24..31, N = lane%16), held in VGPRs for the whole loop.
    const int   nrow = g * H_ + j0 + nt * 16 + ncol;
    const float bias = bconv[nrow];
    v16bf bfrag[4];
    #pragma unroll
    for (int kk = 0; kk < 4; ++kk) {
        const float* wr = Wconv + (size_t)nrow * K_ + kk * 32 + hi * 8;
        v16bf t;
        #pragma unroll
        for (int i = 0; i < 16; ++i) t[i] = (bf16_t)wr[(i & 7) + ((i >> 3) << 4)];
        bfrag[kk] = t;
    }

    float h = 0.0f;                 // recurrent state for j = tid & 63
    const int jloc = tid & 63;
    const int qsel = tid >> 6;      // scan quarter owned by this replica
    const float* Xb = X + (size_t)b * S_ * 8;

    const int tA = tid >> 3;              // comb row this thread builds
    const int k0 = (tid & 7) * 16;        // comb col base
    const int tN = tid >> 2;              // tn row (tid < 128)
    const int oN = tid & 3;               // tn output (tid < 128)

    __syncthreads();

    // ---- stream over time in chunks of 32 steps ------------------------
    for (int ch = 0; ch < NCH; ++ch) {
        const int t0 = ch * TCH;

        // P1) load X chunk + gather embedding rows into comb.
        {
            int t = tid >> 3, c = tid & 7;
            Xc[t][c] = Xb[(size_t)(t0 + t) * 8 + c];

            int ev = ((int)Xb[(size_t)(t0 + tA) * 8]) & (VOCAB_ - 1);
            v8bf*       dst = (v8bf*)&comb[tA][k0];
            const v8bf* src = (const v8bf*)&embS[ev][k0];
            dst[0] = src[0];
            dst[1] = src[1];        // k0==112 copies zero pad into 124..127
        }
        __syncthreads();

        // P2) numeric projection -> comb[t][124..127]
        if (tid < 128) {
            float acc = bnumS[oN];
            #pragma unroll
            for (int i = 0; i < 7; ++i) acc += Xc[tN][1 + i] * wnumS[oN][i];
            comb[tN][EMB_ + oN] = (bf16_t)acc;
        }
        __syncthreads();

        // P3) WMMA GEMM: 16 ds_load_b128 then 8 wmma (pinned by sched
        //     group barriers so waits become partial and latency pipelines).
        v16bf afr[8];
        #pragma unroll
        for (int kk = 0; kk < 4; ++kk) {
            const int kb = kk * 32 + hi * 8;       // A layout K base
            const v8bf* a0p = (const v8bf*)&comb[ncol][kb];
            const v8bf* a1p = (const v8bf*)&comb[16 + ncol][kb];
            v8bf l00 = a0p[0], l01 = a0p[2];       // K kb.., kb+16..
            v8bf l10 = a1p[0], l11 = a1p[2];
            v16bf a0, a1;
            #pragma unroll
            for (int i = 0; i < 8; ++i) {
                a0[i] = l00[i]; a0[i + 8] = l01[i];
                a1[i] = l10[i]; a1[i + 8] = l11[i];
            }
            afr[2 * kk]     = a0;
            afr[2 * kk + 1] = a1;
        }

        v8f c0, c1;
        #pragma unroll
        for (int v = 0; v < 8; ++v) { c0[v] = bias; c1[v] = bias; }
        #pragma unroll
        for (int kk = 0; kk < 4; ++kk) {
            c0 = __builtin_amdgcn_wmma_f32_16x16x32_bf16(
                     false, afr[2 * kk],     false, bfrag[kk], (short)0, c0, false, false);
            c1 = __builtin_amdgcn_wmma_f32_16x16x32_bf16(
                     false, afr[2 * kk + 1], false, bfrag[kk], (short)0, c1, false, false);
        }
        __builtin_amdgcn_sched_group_barrier(0x100, 16, 0);  // 16 x DS read
        __builtin_amdgcn_sched_group_barrier(0x008,  8, 0);  //  8 x WMMA

        // activate + scatter to interleaved gate buffer (t = v + hi*8)
        {
            const int jl2 = 2 * (nt * 16 + ncol) + g;
            #pragma unroll
            for (int v = 0; v < 8; ++v) {
                int t = v + hi * 8;
                float x0 = c0[v], x1 = c1[v], a0v, a1v;
                if (g == 0) { a0v = fast_tanh(x0);    a1v = fast_tanh(x1); }
                else        { a0v = fast_sigmoid(x0); a1v = fast_sigmoid(x1); }
                gZF[t][jl2]      = a0v;
                gZF[t + 16][jl2] = a1v;
            }
        }
        __syncthreads();

        // P4a) each replica quarter composes its 8 affine steps:
        //      (A,C) <- (f*A, z + f*(C - z)),  h' = A*h + C over the window
        {
            float A = 1.0f, C = 0.0f;
            const int tq = qsel * 8;
            #pragma unroll
            for (int t = 0; t < 8; ++t) {
                const float2 zf = *(const float2*)&gZF[tq + t][2 * jloc];
                float z = zf.x, f = zf.y;
                A = A * f;
                C = z + f * (C - z);
            }
            *(float2*)&ACbuf[qsel][jloc][0] = make_float2(A, C);
        }
        __syncthreads();

        // P4b) apply the 4 window maps in time order (4 dependent FMAs)
        #pragma unroll
        for (int q = 0; q < 4; ++q) {
            const float2 ac = *(const float2*)&ACbuf[q][jloc][0];
            h = ac.x * h + ac.y;
        }
        // no barrier needed here: next conflicting LDS write is 3 barriers away
    }

    // ---- output gate at t = S-1 (comb still holds last chunk) ----------
    if (tid < 64) {
        const int n = 2 * H_ + j0 + tid;
        float acc = bconv[n];
        const float* wr = Wconv + (size_t)n * K_;
        #pragma unroll 16
        for (int k = 0; k < K_; ++k) acc += (float)comb[TCH - 1][k] * wr[k];
        float o = fast_sigmoid(acc);
        red[tid] = Wout[j0 + tid] * o * h;
    }
    __syncthreads();
    if (tid == 0) {
        float s = 0.0f;
        #pragma unroll
        for (int i = 0; i < 64; ++i) s += red[i];
        atomicAdd(&out[b], s);
    }
}

__global__ void qrnn_init_out(float* __restrict__ out,
                              const float* __restrict__ b_out) {
    out[threadIdx.x] = b_out[0];
}

extern "C" void kernel_launch(void* const* d_in, const int* in_sizes, int n_in,
                              void* d_out, int out_size, void* d_ws, size_t ws_size,
                              hipStream_t stream) {
    const float* X     = (const float*)d_in[0];
    const float* emb   = (const float*)d_in[1];
    const float* Wnum  = (const float*)d_in[2];
    const float* bnum  = (const float*)d_in[3];
    const float* Wconv = (const float*)d_in[4];
    const float* bconv = (const float*)d_in[5];
    const float* Wout  = (const float*)d_in[6];
    const float* bout  = (const float*)d_in[7];
    float* out = (float*)d_out;

    qrnn_init_out<<<dim3(1), dim3(B_), 0, stream>>>(out, bout);
    qrnn_fused<<<dim3(B_, NJB), dim3(256), 0, stream>>>(
        X, emb, Wnum, bnum, Wconv, bconv, Wout, out);
}
